// SimpleMoE_58377195487789
// MI455X (gfx1250) — matrix operand order repaired and checked by reading, not compiled, not run on hardware
//
#include <hip/hip_runtime.h>
#include <hip/hip_bf16.h>

#define NTOK 32768
#define DDIM 768
#define HDIM 256
#define NEXP 8

typedef _Float16 half_t;
typedef __attribute__((ext_vector_type(16))) _Float16 v16h;
typedef __attribute__((ext_vector_type(8)))  _Float16 v8h;
typedef __attribute__((ext_vector_type(8)))  float    v8f;
typedef __attribute__((ext_vector_type(2)))  float    v2f;

__device__ inline v8f zero8() {
  v8f r;
#pragma unroll
  for (int i = 0; i < 8; ++i) r[i] = 0.0f;
  return r;
}

// ---- WMMA wrappers -------------------------------------------------------
__device__ inline v8f wmma_f16(v16h a, v16h b, v8f c) {
  return __builtin_amdgcn_wmma_f32_16x16x32_f16(false, a, false, b, (short)0, c, false, false);
}
__device__ inline v8f wmma_f32k4(v2f a, v2f b, v8f c) {
  return __builtin_amdgcn_wmma_f32_16x16x4_f32(false, a, false, b, (short)0, c, false, false);
}

// A-fragment (16-bit, 16x32): lane<16 holds row=lane, K={kb..kb+7, kb+16..kb+23};
// lane>=16 holds row=lane-16, K shifted by 8.  (ISA 7.12.2)
__device__ inline v16h fragA_ld(const half_t* base, int ld, int row0, int kb, int lane) {
  const half_t* p = base + (size_t)(row0 + (lane & 15)) * ld + kb + ((lane >> 4) << 3);
  v16h res;
  *((v8h*)&res)     = *(const v8h*)p;
  *((v8h*)&res + 1) = *(const v8h*)(p + 16);
  return res;
}
// B-fragment (16-bit, 32x16) loaded from transposed (NxK row-major) weights:
// lane<16: col=lane, K={kb..kb+7, kb+16..kb+23}; lane>=16: col=lane-16, K shifted by 8.
__device__ inline v16h fragB_ld(const half_t* base, int ld, int n0, int kb, int lane) {
  const half_t* p = base + (size_t)(n0 + (lane & 15)) * ld + kb + ((lane >> 4) << 3);
  v16h res;
  *((v8h*)&res)     = *(const v8h*)p;
  *((v8h*)&res + 1) = *(const v8h*)(p + 16);
  return res;
}

// ---- init / convert / transpose -----------------------------------------
__global__ void k_init(int* counts) {
  if (threadIdx.x < 16) counts[threadIdx.x] = 0;
}

// x f32 -> f16 copy (enables async LDS staging in the expert kernel)
__global__ void k_cvt_x(const float* __restrict__ x, half_t* __restrict__ xh) {
  const long n = (long)NTOK * DDIM;
  for (long i = blockIdx.x * (long)blockDim.x + threadIdx.x; i < n;
       i += (long)gridDim.x * blockDim.x)
    xh[i] = (half_t)x[i];
}

// W1 [E][D][H] f32 -> W1t [E][H][D] f16
__global__ void k_tr1(const float* __restrict__ W1, half_t* __restrict__ W1t) {
  int i = blockIdx.x * blockDim.x + threadIdx.x;
  const int total = NEXP * HDIM * DDIM;
  if (i >= total) return;
  int e = i / (HDIM * DDIM);
  int n = (i / DDIM) % HDIM;
  int k = i % DDIM;
  W1t[i] = (half_t)W1[(size_t)e * DDIM * HDIM + (size_t)k * HDIM + n];
}
// W2 [E][H][D] f32 -> W2t [E][D][H] f16
__global__ void k_tr2(const float* __restrict__ W2, half_t* __restrict__ W2t) {
  int i = blockIdx.x * blockDim.x + threadIdx.x;
  const int total = NEXP * DDIM * HDIM;
  if (i >= total) return;
  int e = i / (DDIM * HDIM);
  int n = (i / HDIM) % DDIM;
  int k = i % HDIM;
  W2t[i] = (half_t)W2[(size_t)e * HDIM * DDIM + (size_t)k * DDIM + n];
}

// ---- gate ----------------------------------------------------------------
// 16 tokens per block, 256 threads (8 waves). f32 WMMA for x@Wg1 (exact fp32).
__global__ void k_gate(const float* __restrict__ x,  const float* __restrict__ Wg1,
                       const float* __restrict__ bg1, const float* __restrict__ Wg2,
                       const float* __restrict__ bg2, float* __restrict__ gate_out,
                       int* __restrict__ counts, int* __restrict__ lists,
                       float* __restrict__ probs) {
  __shared__ float xs[16 * DDIM];   // 48KB; reused as hidden[16][256] later
  __shared__ float sc[16 * NEXP];
  const int tid = threadIdx.x;
  const int tok0 = blockIdx.x * 16;

  for (int i = tid; i < 16 * DDIM; i += 256)
    xs[i] = x[(size_t)(tok0 + i / DDIM) * DDIM + (i % DDIM)];
  __syncthreads();

  const int lane = tid & 31, w = tid >> 5;
  const int row = lane & 15, hi = lane >> 4;
  const int klo = hi * 2;
  v8f acc[2];
  acc[0] = zero8();
  acc[1] = zero8();

  for (int kb = 0; kb < DDIM; kb += 4) {
    v2f a; a[0] = xs[row * DDIM + kb + klo];
           a[1] = xs[row * DDIM + kb + klo + 1];
#pragma unroll
    for (int t = 0; t < 2; ++t) {
      const int n0 = (w * 2 + t) * 16;
      v2f b; b[0] = Wg1[(size_t)(kb + klo)     * HDIM + n0 + row];
             b[1] = Wg1[(size_t)(kb + klo + 1) * HDIM + n0 + row];
      acc[t] = wmma_f32k4(a, b, acc[t]);
    }
  }
  __syncthreads();  // everyone is done reading xs

  // hidden (relu) into xs[0 .. 4096)
#pragma unroll
  for (int t = 0; t < 2; ++t) {
    const int col = (w * 2 + t) * 16 + row;
    const float bias = bg1[col];
#pragma unroll
    for (int j = 0; j < 8; ++j)
      xs[(j + 8 * hi) * HDIM + col] = fmaxf(acc[t][j] + bias, 0.0f);
  }
  __syncthreads();

  if (tid < 16 * NEXP) {
    const int t = tid >> 3, e = tid & 7;
    float s = bg2[e];
    for (int k = 0; k < HDIM; ++k) s += xs[t * HDIM + k] * Wg2[(size_t)k * NEXP + e];
    sc[t * NEXP + e] = s;
  }
  __syncthreads();

  if (tid < 16) {
    float p[NEXP], m = -1e30f;
#pragma unroll
    for (int e = 0; e < NEXP; ++e) { p[e] = sc[tid * NEXP + e]; m = fmaxf(m, p[e]); }
    float sum = 0.0f;
#pragma unroll
    for (int e = 0; e < NEXP; ++e) { p[e] = expf(p[e] - m); sum += p[e]; }
    const float inv = 1.0f / sum;
    const int tokg = tok0 + tid;
    int i1 = 0; float p1 = -1.0f;
#pragma unroll
    for (int e = 0; e < NEXP; ++e) {
      const float pe = p[e] * inv;
      gate_out[(size_t)tokg * NEXP + e] = pe;
      if (pe > p1) { p1 = pe; i1 = e; }
    }
    int i2 = 0; float p2 = -1.0f;
#pragma unroll
    for (int e = 0; e < NEXP; ++e) {
      if (e == i1) continue;
      const float pe = p[e] * inv;
      if (pe > p2) { p2 = pe; i2 = e; }
    }
    const float s12 = p1 + p2;
    int pos = atomicAdd(&counts[i1], 1);
    lists[i1 * NTOK + pos] = tokg;                 // slot 0
    probs[i1 * NTOK + pos] = p1 / s12;
    pos = atomicAdd(&counts[i2], 1);
    lists[i2 * NTOK + pos] = tokg | (1 << 16);     // slot 1
    probs[i2 * NTOK + pos] = p2 / s12;
  }
}

// ---- expert FFN (f16 WMMA) ----------------------------------------------
// grid = (E, ceil(N/32)); block = 256 threads = 8 waves; 32 tokens per block.
__global__ void k_expert(const float* __restrict__ x, const half_t* __restrict__ xh,
                         const half_t* __restrict__ W1t, const float* __restrict__ b1,
                         const half_t* __restrict__ W2t, const float* __restrict__ b2,
                         const int* __restrict__ counts, const int* __restrict__ lists,
                         const float* __restrict__ probs,
                         float* __restrict__ slotbuf, float* __restrict__ out,
                         int useXh, int slotMode) {
  const int e = blockIdx.x;
  const int chunk = blockIdx.y;
  const int cnt = counts[e];
  if (chunk * 32 >= cnt) return;

  __shared__ half_t smem[32 * DDIM];  // 48KB: x tile (f16); later h tile [32][256]
  __shared__ int   sTok[32];
  __shared__ int   sSlot[32];
  __shared__ float sProb[32];

  const int tid = threadIdx.x;
  if (tid < 32) {
    const int idx = chunk * 32 + tid;
    const int cl = idx < cnt ? idx : cnt - 1;
    const int packed = lists[e * NTOK + cl];
    sTok[tid] = packed & 0xFFFF;
    sSlot[tid] = packed >> 16;
    sProb[tid] = (idx < cnt) ? probs[e * NTOK + cl] : 0.0f;
  }
  __syncthreads();

  const half_t* W1e = W1t + (size_t)e * HDIM * DDIM;
  const half_t* W2e = W2t + (size_t)e * DDIM * HDIM;
  __builtin_prefetch(W1e, 0, 1);
  __builtin_prefetch(W2e, 0, 1);

  if (useXh) {
    // Async gather: global f16 rows -> LDS, 16B per lane per issue (ASYNCcnt).
#pragma unroll
    for (int it = 0; it < (32 * DDIM) / (256 * 8); ++it) {   // 12 iterations
      const int idx = it * 256 + tid;                        // 16B chunk index
      const int r  = idx / (DDIM / 8);
      const int c  = (idx % (DDIM / 8)) * 8;
      const half_t* src = xh + (size_t)sTok[r] * DDIM + c;
      const unsigned dst = (unsigned)(size_t)&smem[r * DDIM + c];
      asm volatile("global_load_async_to_lds_b128 %0, %1, off"
                   :: "v"(dst), "v"(src) : "memory");
    }
    asm volatile("s_wait_asynccnt 0x0" ::: "memory");
  } else {
    for (int i = tid; i < 32 * DDIM; i += 256) {
      const int r = i / DDIM, c = i % DDIM;
      smem[i] = (half_t)x[(size_t)sTok[r] * DDIM + c];
    }
  }
  __syncthreads();

  const int lane = tid & 31, w = tid >> 5;
  const int rb = w & 1, cg = w >> 1;
  const int m0 = rb * 16;
  const int nlo = lane & 15, hi = lane >> 4;

  // layer 1: h[32x256] = relu(x @ W1e + b1)
  v8f acc[4];
#pragma unroll
  for (int t = 0; t < 4; ++t) acc[t] = zero8();
  for (int ks = 0; ks < DDIM / 32; ++ks) {
    const v16h a = fragA_ld(smem, DDIM, m0, ks * 32, lane);
#pragma unroll
    for (int t = 0; t < 4; ++t) {
      const int n0 = (cg * 4 + t) * 16;
      const v16h b = fragB_ld(W1e, DDIM, n0, ks * 32, lane);
      acc[t] = wmma_f16(a, b, acc[t]);
    }
  }
  __syncthreads();  // x tile fully consumed

#pragma unroll
  for (int t = 0; t < 4; ++t) {
    const int col = (cg * 4 + t) * 16 + nlo;
    const float bias = b1[e * HDIM + col];
#pragma unroll
    for (int j = 0; j < 8; ++j) {
      const int r = m0 + j + 8 * hi;
      smem[r * HDIM + col] = (half_t)fmaxf(acc[t][j] + bias, 0.0f);
    }
  }
  __syncthreads();

  // layer 2: y[32x768] = h @ W2e + b2 ; write p*y
  v8f acc2[12];
#pragma unroll
  for (int t = 0; t < 12; ++t) acc2[t] = zero8();
  for (int ks = 0; ks < HDIM / 32; ++ks) {
    const v16h a = fragA_ld(smem, HDIM, m0, ks * 32, lane);
#pragma unroll
    for (int t = 0; t < 12; ++t) {
      const int n0 = (cg * 12 + t) * 16;
      const v16h b = fragB_ld(W2e, HDIM, n0, ks * 32, lane);
      acc2[t] = wmma_f16(a, b, acc2[t]);
    }
  }

  const int base_i = chunk * 32;
#pragma unroll
  for (int t = 0; t < 12; ++t) {
    const int col = (cg * 12 + t) * 16 + nlo;
    const float bias = b2[e * DDIM + col];
#pragma unroll
    for (int j = 0; j < 8; ++j) {
      const int r = m0 + j + 8 * hi;
      if (base_i + r < cnt) {
        const float val = sProb[r] * (acc2[t][j] + bias);
        const size_t o = (size_t)sTok[r] * DDIM + col;
        if (slotMode) slotbuf[(size_t)sSlot[r] * NTOK * DDIM + o] = val;
        else          atomicAdd(&out[o], val);
      }
    }
  }
}

// ---- epilogue helpers ----------------------------------------------------
__global__ void k_zero(float* __restrict__ out, long n) {
  for (long i = blockIdx.x * (long)blockDim.x + threadIdx.x; i < n;
       i += (long)gridDim.x * blockDim.x) out[i] = 0.0f;
}
__global__ void k_combine(const float* __restrict__ s0, const float* __restrict__ s1,
                          float* __restrict__ out, long n) {
  for (long i = blockIdx.x * (long)blockDim.x + threadIdx.x; i < n;
       i += (long)gridDim.x * blockDim.x) out[i] = s0[i] + s1[i];
}

extern "C" void kernel_launch(void* const* d_in, const int* in_sizes, int n_in,
                              void* d_out, int out_size, void* d_ws, size_t ws_size,
                              hipStream_t stream) {
  const float* x   = (const float*)d_in[0];
  const float* W1  = (const float*)d_in[1];
  const float* b1  = (const float*)d_in[2];
  const float* W2  = (const float*)d_in[3];
  const float* b2  = (const float*)d_in[4];
  const float* Wg1 = (const float*)d_in[5];
  const float* bg1 = (const float*)d_in[6];
  const float* Wg2 = (const float*)d_in[7];
  const float* bg2 = (const float*)d_in[8];

  float* out = (float*)d_out;
  float* gate_out = out + (size_t)NTOK * DDIM;

  char* ws = (char*)d_ws;
  size_t off = 0;
  half_t* W1t = (half_t*)(ws + off); off += (size_t)NEXP * HDIM * DDIM * 2;
  half_t* W2t = (half_t*)(ws + off); off += (size_t)NEXP * DDIM * HDIM * 2;
  off = (off + 255) & ~(size_t)255;
  int*   counts = (int*)(ws + off);   off += 256;
  int*   lists  = (int*)(ws + off);   off += (size_t)NEXP * NTOK * 4;
  float* probs  = (float*)(ws + off); off += (size_t)NEXP * NTOK * 4;
  off = (off + 255) & ~(size_t)255;
  half_t* xh = (half_t*)(ws + off);   off += (size_t)NTOK * DDIM * 2;
  const int useXh = (ws_size >= off) ? 1 : 0;
  off = (off + 255) & ~(size_t)255;
  float* slotbuf = (float*)(ws + off);
  const size_t need = off + (size_t)2 * NTOK * DDIM * 4;
  const int slotMode = (ws_size >= need) ? 1 : 0;

  k_init<<<1, 32, 0, stream>>>(counts);
  {
    const int total = NEXP * HDIM * DDIM;
    k_tr1<<<(total + 255) / 256, 256, 0, stream>>>(W1, W1t);
    k_tr2<<<(total + 255) / 256, 256, 0, stream>>>(W2, W2t);
  }
  if (useXh)
    k_cvt_x<<<4096, 256, 0, stream>>>(x, xh);
  k_gate<<<NTOK / 16, 256, 0, stream>>>(x, Wg1, bg1, Wg2, bg2, gate_out,
                                        counts, lists, probs);
  if (!slotMode)
    k_zero<<<4096, 256, 0, stream>>>(out, (long)NTOK * DDIM);
  k_expert<<<dim3(NEXP, NTOK / 32), 256, 0, stream>>>(
      x, xh, W1t, b1, W2t, b2, counts, lists, probs, slotbuf, out,
      useXh, slotMode);
  if (slotMode)
    k_combine<<<8192, 256, 0, stream>>>(slotbuf, slotbuf + (size_t)NTOK * DDIM,
                                        out, (long)NTOK * DDIM);
}
